// New_Audio_Guided_Attention_11828339933296
// MI455X (gfx1250) — matrix operand order repaired and checked by reading, not compiled, not run on hardware
//
#include <hip/hip_runtime.h>
#include <hip/hip_bf16.h>

typedef __bf16 v16bf __attribute__((ext_vector_type(16)));
typedef __bf16 bf16x8 __attribute__((ext_vector_type(8)));
typedef float  v8f   __attribute__((ext_vector_type(8)));

#define LDA 520   // bf16 stride of 64x512 activation tile (16B-aligned rows)
#define LDC 264   // f32 stride of 64x256 cq tile
#define T_DIM 10
#define B_DIM 256

#define WV1_ELEMS (512 * 512)
#define WV2_ELEMS (256 * 512)
#define WS_NEEDED ((size_t)(WV1_ELEMS + WV2_ELEMS) * sizeof(__bf16))

// ---------------------------------------------------------------------------
// Prepass: convert Wv1 (512x512) and Wv2 (256x512) f32 -> bf16 into d_ws.
// Runs once per launch; removes per-workgroup f32->bf16 conversion from the
// WMMA hot loop and halves the L2 weight stream.
// ---------------------------------------------------------------------------
__global__ __launch_bounds__(256)
void convert_weights(const float* __restrict__ Wv1, const float* __restrict__ Wv2,
                     __bf16* __restrict__ dst) {
    const int i = blockIdx.x * 256 + threadIdx.x;
    if (i < WV1_ELEMS) dst[i] = (__bf16)Wv1[i];
    if (i < WV2_ELEMS) dst[WV1_ELEMS + i] = (__bf16)Wv2[i];
}

// A fragment: 16-bit A 16x32 (MxK). Lane m (0-15); lanes 16-31 add +16 to K.
// slots 0..15 -> K = kbase + khi + 0..15
__device__ __forceinline__ v16bf load_a_frag(const __bf16* base, int mtile, int kbase, int lane) {
    const int m   = (lane & 15) + (mtile << 4);
    const int khi = (lane >> 4) << 4;            // 0 or 16
    const __bf16* p = base + m * LDA + kbase + khi;
    bf16x8 lo = *(const bf16x8*)(p);
    bf16x8 hi = *(const bf16x8*)(p + 8);
    v16bf a;
#pragma unroll
    for (int i = 0; i < 8; ++i) { a[i] = lo[i]; a[8 + i] = hi[i]; }
    return a;
}

// B fragment: 16-bit B 32x16 (KxN), W row-major [N][ldw], B[k][n] = W[n][k].
// lanes 0-15: N=lane, K=kbase+0..15 ; lanes 16-31: N=lane-16, K=kbase+16..31
template <bool BF16W>
__device__ __forceinline__ v16bf load_b_frag(const void* __restrict__ W, int ldw,
                                             int ntile, int kbase, int lane) {
    const int n  = (lane & 15) + (ntile << 4);
    const int kb = kbase + ((lane >> 4) << 4);
    v16bf b;
    if constexpr (BF16W) {
        const __bf16* p = (const __bf16*)W + (size_t)n * ldw + kb;
        bf16x8 lo = *(const bf16x8*)(p);     // 16B global_load_b128
        bf16x8 hi = *(const bf16x8*)(p + 8); // 16B global_load_b128
#pragma unroll
        for (int i = 0; i < 8; ++i) { b[i] = lo[i]; b[8 + i] = hi[i]; }
    } else {
        const float* p = (const float*)W + (size_t)n * ldw + kb;
#pragma unroll
        for (int i = 0; i < 16; ++i) b[i] = (__bf16)p[i];
    }
    return b;
}

template <bool BF16W>
__global__ __launch_bounds__(512)
void agatt_fused(const float* __restrict__ video, const float* __restrict__ audio,
                 const void* __restrict__ Wv1g, const float* __restrict__ bv1,
                 const float* __restrict__ Wa1, const float* __restrict__ ba1,
                 const float* __restrict__ Wb,  const float* __restrict__ bb,
                 const float* __restrict__ Wc,  const float* __restrict__ bc,
                 const void* __restrict__ Wv2g, const float* __restrict__ bv2,
                 const float* __restrict__ Wa2, const float* __restrict__ ba2,
                 const float* __restrict__ Ws,  const float* __restrict__ bs,
                 float* __restrict__ out)
{
    __shared__ __bf16 vfb[64 * LDA];     // bf16 activation tile (vf, later c_att)
    __shared__ float  cq [64 * LDC];     // GEMM2 output
    __shared__ float  afs[128];
    __shared__ float  aq1s[512];
    __shared__ float  colacc[512];
    __shared__ float  avqs[512];
    __shared__ float  y1s[256];
    __shared__ float  cam[512];
    __shared__ float  wvec[256];
    __shared__ float  satt[64];

    const int tid  = threadIdx.x;
    const int lane = tid & 31;
    const int wave = tid >> 5;
    const int bt   = blockIdx.x;
    const int b    = bt / T_DIM;
    const int t    = bt - b * T_DIM;

    // ---------------- phase 0: stage audio feature + bf16 visual tile ----------------
    if (tid < 128) afs[tid] = audio[((size_t)t * B_DIM + b) * 128 + tid];
    colacc[tid] = 0.0f;
    {
        const float4* src = (const float4*)(video + (size_t)bt * 49 * 512);
        for (int i = tid; i < 49 * 128; i += 512) {     // 49 rows * 128 float4
            float4 v = src[i];
            const int row = i >> 7;
            const int col = (i & 127) << 2;
            __bf16* d = &vfb[row * LDA + col];
            d[0] = (__bf16)v.x; d[1] = (__bf16)v.y; d[2] = (__bf16)v.z; d[3] = (__bf16)v.w;
        }
        for (int i = tid; i < 15 * LDA; i += 512)       // zero pad rows 49..63
            vfb[49 * LDA + i] = (__bf16)0.0f;
    }
    __syncthreads();

    // ---------------- phase 1: aq1 / wvec, and GEMM1 (vq1) via WMMA ----------------
    {
        float s = ba1[tid];
        const float* w = Wa1 + (size_t)tid * 128;
        for (int k = 0; k < 128; ++k) s += afs[k] * w[k];
        aq1s[tid] = fmaxf(s, 0.0f);
    }
    if (tid < 256) {
        float s = ba2[tid];
        const float* w = Wa2 + (size_t)tid * 128;
        for (int k = 0; k < 128; ++k) s += afs[k] * w[k];
        wvec[tid] = fmaxf(s, 0.0f) * Ws[tid];           // aq2 * Ws fused
    }

    // GEMM1: D[64,512] = vfb[64,512] @ Wv1^T ; 32 N-tiles, 2 per wave
    for (int nt = wave; nt < 32; nt += 16) {
        v8f acc0 = {}, acc1 = {}, acc2 = {}, acc3 = {};
        for (int kk = 0; kk < 16; ++kk) {
            const int kbase = kk << 5;
            v16bf bf = load_b_frag<BF16W>(Wv1g, 512, nt, kbase, lane);
            v16bf a0 = load_a_frag(vfb, 0, kbase, lane);
            v16bf a1 = load_a_frag(vfb, 1, kbase, lane);
            v16bf a2 = load_a_frag(vfb, 2, kbase, lane);
            v16bf a3 = load_a_frag(vfb, 3, kbase, lane);
            acc0 = __builtin_amdgcn_wmma_f32_16x16x32_bf16(false, a0, false, bf, (short)0, acc0, false, false);
            acc1 = __builtin_amdgcn_wmma_f32_16x16x32_bf16(false, a1, false, bf, (short)0, acc1, false, false);
            acc2 = __builtin_amdgcn_wmma_f32_16x16x32_bf16(false, a2, false, bf, (short)0, acc2, false, false);
            acc3 = __builtin_amdgcn_wmma_f32_16x16x32_bf16(false, a3, false, bf, (short)0, acc3, false, false);
        }
        const int n     = (lane & 15) + (nt << 4);
        const int rbase = (lane >> 4) << 3;             // M = v + rbase (per C/D layout)
        const float bias = bv1[n];
        float partial = 0.0f;
#pragma unroll
        for (int v = 0; v < 8; ++v) {
            float x;
            x = acc0[v] + bias; if (x > 0.0f && (rbase + v)      < 49) partial += x;
            x = acc1[v] + bias; if (x > 0.0f && (16 + rbase + v) < 49) partial += x;
            x = acc2[v] + bias; if (x > 0.0f && (32 + rbase + v) < 49) partial += x;
            x = acc3[v] + bias; if (x > 0.0f && (48 + rbase + v) < 49) partial += x;
        }
        atomicAdd(&colacc[n], partial);                 // ds_add_f32
    }
    __syncthreads();

    // ---------------- phase 2: channel attention MLP ----------------
    avqs[tid] = aq1s[tid] * colacc[tid] * (1.0f / 49.0f);
    __syncthreads();
    if (tid < 256) {
        float s = bb[tid];
        const float* w = Wb + (size_t)tid * 512;
        for (int k = 0; k < 512; ++k) s += avqs[k] * w[k];
        y1s[tid] = fmaxf(s, 0.0f);
    }
    __syncthreads();
    {
        float s = bc[tid];
        const float* w = Wc + (size_t)tid * 256;
        for (int k = 0; k < 256; ++k) s += y1s[k] * w[k];
        cam[tid] = 1.0f + 1.0f / (1.0f + __expf(-s));   // sigmoid + 1
    }
    __syncthreads();

    // scale tile in place: c_att = vf * cam   (pad rows stay zero)
    for (int i = tid; i < 64 * 512; i += 512) {
        const int row = i >> 9, col = i & 511;
        const int idx = row * LDA + col;
        vfb[idx] = (__bf16)((float)vfb[idx] * cam[col]);
    }
    __syncthreads();

    // ---------------- phase 3: GEMM2 (cq) via WMMA, 16 N-tiles, 1 per wave ----------
    {
        const int nt = wave;
        v8f acc0 = {}, acc1 = {}, acc2 = {}, acc3 = {};
        for (int kk = 0; kk < 16; ++kk) {
            const int kbase = kk << 5;
            v16bf bf = load_b_frag<BF16W>(Wv2g, 512, nt, kbase, lane);
            v16bf a0 = load_a_frag(vfb, 0, kbase, lane);
            v16bf a1 = load_a_frag(vfb, 1, kbase, lane);
            v16bf a2 = load_a_frag(vfb, 2, kbase, lane);
            v16bf a3 = load_a_frag(vfb, 3, kbase, lane);
            acc0 = __builtin_amdgcn_wmma_f32_16x16x32_bf16(false, a0, false, bf, (short)0, acc0, false, false);
            acc1 = __builtin_amdgcn_wmma_f32_16x16x32_bf16(false, a1, false, bf, (short)0, acc1, false, false);
            acc2 = __builtin_amdgcn_wmma_f32_16x16x32_bf16(false, a2, false, bf, (short)0, acc2, false, false);
            acc3 = __builtin_amdgcn_wmma_f32_16x16x32_bf16(false, a3, false, bf, (short)0, acc3, false, false);
        }
        const int n     = (lane & 15) + (nt << 4);
        const int rbase = (lane >> 4) << 3;
        const float bias = bv2[n];
#pragma unroll
        for (int v = 0; v < 8; ++v) {
            cq[(rbase + v)      * LDC + n] = fmaxf(acc0[v] + bias, 0.0f);
            cq[(16 + rbase + v) * LDC + n] = fmaxf(acc1[v] + bias, 0.0f);
            cq[(32 + rbase + v) * LDC + n] = fmaxf(acc2[v] + bias, 0.0f);
            cq[(48 + rbase + v) * LDC + n] = fmaxf(acc3[v] + bias, 0.0f);
        }
    }
    __syncthreads();

    // ---------------- phase 4: spatial attention ----------------
    if (tid < 49) {
        float s = bs[0];
        const float* row = &cq[tid * LDC];
        for (int k = 0; k < 256; ++k) s += row[k] * wvec[k];
        satt[tid] = tanhf(s);
    }
    __syncthreads();
    if (tid == 0) {
        float mx = -1e30f;
        for (int i = 0; i < 49; ++i) mx = fmaxf(mx, satt[i]);
        float sum = 0.0f;
        for (int i = 0; i < 49; ++i) { float e = __expf(satt[i] - mx); satt[i] = e; sum += e; }
        const float inv = 1.0f / sum;
        for (int i = 0; i < 49; ++i) satt[i] *= inv;
    }
    __syncthreads();

    // ---------------- phase 5: out = s_att @ c_att ----------------
    {
        float o = 0.0f;
        for (int hw = 0; hw < 49; ++hw)
            o += satt[hw] * (float)vfb[hw * LDA + tid];
        out[(size_t)bt * 512 + tid] = o;
    }
}

extern "C" void kernel_launch(void* const* d_in, const int* in_sizes, int n_in,
                              void* d_out, int out_size, void* d_ws, size_t ws_size,
                              hipStream_t stream) {
    const float* video = (const float*)d_in[0];
    const float* audio = (const float*)d_in[1];
    const float* Wv1 = (const float*)d_in[2];
    const float* bv1 = (const float*)d_in[3];
    const float* Wa1 = (const float*)d_in[4];
    const float* ba1 = (const float*)d_in[5];
    const float* Wb  = (const float*)d_in[6];
    const float* bb  = (const float*)d_in[7];
    const float* Wc  = (const float*)d_in[8];
    const float* bc  = (const float*)d_in[9];
    const float* Wv2 = (const float*)d_in[10];
    const float* bv2 = (const float*)d_in[11];
    const float* Wa2 = (const float*)d_in[12];
    const float* ba2 = (const float*)d_in[13];
    const float* Ws  = (const float*)d_in[14];
    const float* bs  = (const float*)d_in[15];
    float* out = (float*)d_out;

    const int BT = B_DIM * T_DIM;   // 2560 workgroups, one per (b,t)

    if (ws_size >= WS_NEEDED) {
        // Prepass: bf16 weight images in scratch (re-done every call; deterministic).
        __bf16* wb = (__bf16*)d_ws;
        convert_weights<<<(WV1_ELEMS + 255) / 256, 256, 0, stream>>>(Wv1, Wv2, wb);
        const __bf16* wv1b = wb;
        const __bf16* wv2b = wb + WV1_ELEMS;
        agatt_fused<true><<<BT, 512, 0, stream>>>(video, audio, wv1b, bv1, Wa1, ba1,
                                                  Wb, bb, Wc, bc, wv2b, bv2,
                                                  Wa2, ba2, Ws, bs, out);
    } else {
        // Fallback: convert f32 weights in-loop (same math, more VALU).
        agatt_fused<false><<<BT, 512, 0, stream>>>(video, audio, Wv1, bv1, Wa1, ba1,
                                                   Wb, bb, Wc, bc, Wv2, bv2,
                                                   Wa2, ba2, Ws, bs, out);
    }
}
